// E23Cell_8151847928499
// MI455X (gfx1250) — compile-verified, matrix operand order checked by reading, not run on hardware
//
#include <hip/hip_runtime.h>
#include <math.h>

#define B_ 16
#define T_ 4096
#define D_ 512
#define S_ 64

typedef __attribute__((ext_vector_type(16))) __bf16 v16bf;
typedef __attribute__((ext_vector_type(2)))  __bf16 v2bf;
typedef __attribute__((ext_vector_type(8)))  float  v8f;

// ---------- helpers (native bf16 converts -> v_cvt_pk_bf16_f32) ----------
__device__ __forceinline__ unsigned short f2bf(float f) {
    __bf16 h = (__bf16)f;
    return __builtin_bit_cast(unsigned short, h);
}
__device__ __forceinline__ unsigned pack2bf(float a, float b) {
    v2bf v;
    v.x = (__bf16)a;
    v.y = (__bf16)b;
    return __builtin_bit_cast(unsigned, v);
}

union FragB { v16bf v; uint4 q[2]; unsigned u[8]; };

// B-fragment (32x16 K x N, bf16) from fragment-major packed weights.
// word index = ((kk*512 + n)*2 + khalf)*8 + v  -> each lane: 8 contiguous dwords.
__device__ __forceinline__ v16bf load_frag_b_fm(const unsigned* P, int kk, int nb, int lane) {
    size_t w = ((((size_t)(kk << 9) + nb + (lane & 15)) << 1) + (lane >> 4)) << 3;
    const uint4* q = (const uint4*)(P + w);
    FragB f;
    f.q[0] = q[0];
    f.q[1] = q[1];
    return f.v;
}

// ---------- kernel 1: pack a 512x512 fp32 weight (W[d][k], used as h@W^T)
// into fragment-major bf16 B-operand layout ----------
__global__ void pack_weight_fm(const float* __restrict__ W, unsigned* __restrict__ P) {
    int idx = blockIdx.x * blockDim.x + threadIdx.x;   // 0 .. 131071
    int v    = idx & 7;
    int half = (idx >> 3) & 1;
    int n    = (idx >> 4) & 511;
    int kk   = idx >> 13;
    int k = (kk << 5) + (half << 3) + ((v >> 2) << 4) + ((v & 3) << 1);
    // B[k][n] = W[n][k]
    float w0 = W[n * 512 + k];
    float w1 = W[n * 512 + k + 1];
    P[idx] = pack2bf(w0, w1);
}

// ---------- kernel 2: xw = x @ W_x^T + b_h  (65536 x 512 x 512, WMMA bf16)
// Each wave: 2 M-tiles x 1 N-tile -> one B fragment feeds two independent WMMAs.
__global__ __launch_bounds__(256) void xw_gemm(const float* __restrict__ x,
                                               const unsigned* __restrict__ WxP,
                                               const float* __restrict__ bh,
                                               float* __restrict__ out0) {
    int lane = threadIdx.x & 31;
    int w    = threadIdx.x >> 5;
    int jt   = blockIdx.y * 8 + w;          // N tile 0..31
    int nb   = jt << 4;
    int mbase = blockIdx.x << 5;            // 2 M tiles (rows = b*T + t)
    int n = nb + (lane & 15);
    int mh = (lane >> 4) << 3;

    float bias = bh[n];
    v8f c0, c1;
#pragma unroll
    for (int r = 0; r < 8; ++r) { c0[r] = bias; c1[r] = bias; }

    int m0 = mbase + (lane & 15);
    int m1 = m0 + 16;
#pragma unroll
    for (int kk = 0; kk < 16; ++kk) {
        FragB a0, a1;
#pragma unroll
        for (int v = 0; v < 8; ++v) {
            int k = (kk << 5) + ((v >= 4) ? 16 : 0) + mh + ((v & 3) << 1);
            float2 f0 = *(const float2*)(x + (size_t)m0 * 512 + k);
            float2 f1 = *(const float2*)(x + (size_t)m1 * 512 + k);
            a0.u[v] = pack2bf(f0.x, f0.y);
            a1.u[v] = pack2bf(f1.x, f1.y);
        }
        v16bf b = load_frag_b_fm(WxP, kk, nb, lane);
        c0 = __builtin_amdgcn_wmma_f32_16x16x32_bf16(false, a0.v, false, b,
                                                     (short)0, c0, false, false);
        c1 = __builtin_amdgcn_wmma_f32_16x16x32_bf16(false, a1.v, false, b,
                                                     (short)0, c1, false, false);
    }
#pragma unroll
    for (int r = 0; r < 8; ++r) {
        int mr = mbase + r + mh;
        out0[((size_t)mr << 9) + n] = c0[r];
        out0[((size_t)(mr + 16) << 9) + n] = c1[r];
    }
}

// ---------- kernel 3: flat copy (tape init) ----------
__global__ void copy_f32(const float* __restrict__ src, float* __restrict__ dst, int nelem) {
    int i = blockIdx.x * blockDim.x + threadIdx.x;
    if (i < nelem) dst[i] = src[i];
}

// ---------- kernel 4: the sequential scan (single workgroup, 16 waves) ----------
__global__ __launch_bounds__(512) void recurrence(const float* __restrict__ h0,
                                                  const unsigned* __restrict__ WhP,
                                                  const unsigned* __restrict__ WwP,
                                                  float* __restrict__ out0,   // in: xw, out: h_seq
                                                  float* __restrict__ tape) { // [16][64][512] fp32
    __shared__ unsigned short hFrag[16 * 256 * 2]; // bf16 h in A-fragment-major, 16 KB
    __shared__ float readv[16 * 512];              // attention read vector, 32 KB
    __shared__ float sc[16 * 64];                  // scores / attn, 4 KB

    const int tid  = threadIdx.x;
    const int lane = tid & 31;
    const int wv   = tid >> 5;

    // init h (bf16, fragment-major) from h_work
    for (int i = tid; i < 16 * 512; i += 512) {
        int b = i >> 9, k = i & 511;
        int kk = k >> 5, half = (k >> 3) & 1;
        int v = ((k >> 4) & 1) * 4 + ((k >> 1) & 3);
        int widx = (kk << 8) + ((b + (half << 4)) << 3) + v;
        hFrag[(widx << 1) + (k & 1)] = f2bf(h0[i]);
    }
    __syncthreads();

    const int n0  = wv << 5;        // this wave owns 32 output columns (2 N tiles)
    const int nb0 = n0;
    const int nb1 = n0 + 16;
    const int mh  = (lane >> 4) << 3;
    const int nA  = nb0 + (lane & 15);
    const int nBt = nb1 + (lane & 15);

    for (int t = 0; t < T_; ++t) {
        // ---- phase 1: scores[b][s] = (tape[b,s,:] . h[b]) / sqrt(D)
#pragma unroll
        for (int pp = 0; pp < 2; ++pp) {
            int p = tid + (pp << 9);
            int b = p >> 6, s = p & 63;
            const float* trow = tape + ((size_t)((b << 6) + s) << 9);
            float acc = 0.f;
            for (int kk = 0; kk < 16; ++kk) {
#pragma unroll
                for (int half = 0; half < 2; ++half) {
                    int base = ((kk << 8) + ((b + (half << 4)) << 3)) << 1;
#pragma unroll
                    for (int v = 0; v < 8; ++v) {
                        unsigned u = *(const unsigned*)&hFrag[base + (v << 1)];
                        int k = (kk << 5) + (half << 3) + ((v >> 2) << 4) + ((v & 3) << 1);
                        float2 tv = *(const float2*)(trow + k);
                        acc += __uint_as_float(u << 16) * tv.x +
                               __uint_as_float(u & 0xffff0000u) * tv.y;
                    }
                }
            }
            sc[(b << 6) + s] = acc * 0.044194173824159216f; // 1/sqrt(512)
        }
        __syncthreads();

        // ---- softmax per batch (16 lanes do 16 batches)
        if (tid < 16) {
            float mx = -1e30f;
            for (int s = 0; s < 64; ++s) mx = fmaxf(mx, sc[(tid << 6) + s]);
            float sum = 0.f;
            for (int s = 0; s < 64; ++s) {
                float e = expf(sc[(tid << 6) + s] - mx);
                sc[(tid << 6) + s] = e;
                sum += e;
            }
            float inv = 1.f / sum;
            for (int s = 0; s < 64; ++s) sc[(tid << 6) + s] *= inv;
        }
        __syncthreads();

        // ---- phase 1b: read[b][d] = sum_s attn[b][s] * tape[b][s][d]
        {
            int b = tid >> 5, dbase = (tid & 31) << 4;
            float r[16];
#pragma unroll
            for (int j = 0; j < 16; ++j) r[j] = 0.f;
            for (int s = 0; s < 64; ++s) {
                float a = sc[(b << 6) + s];
                const float4* tr = (const float4*)(tape + ((size_t)((b << 6) + s) << 9) + dbase);
#pragma unroll
                for (int q = 0; q < 4; ++q) {
                    float4 f = tr[q];
                    r[q * 4 + 0] += a * f.x;
                    r[q * 4 + 1] += a * f.y;
                    r[q * 4 + 2] += a * f.z;
                    r[q * 4 + 3] += a * f.w;
                }
            }
#pragma unroll
            for (int j = 0; j < 16; ++j) readv[(b << 9) + dbase + j] = r[j];
        }
        __syncthreads();

        // ---- phase 2: h_new = tanh(h @ Wh^T + xw[t] + read)  (WMMA bf16)
        v16bf afr[16];
#pragma unroll
        for (int kk = 0; kk < 16; ++kk) {
            FragB f;
            const uint4* q = (const uint4*)&hFrag[((kk << 8) + (lane << 3)) << 1];
            f.q[0] = q[0];
            f.q[1] = q[1];
            afr[kk] = f.v;
        }
        __syncthreads(); // all A-fragments captured before hFrag is overwritten

        {
            v8f c0, c1;
#pragma unroll
            for (int r = 0; r < 8; ++r) {
                int m = r + mh; // batch
                c0[r] = out0[(((size_t)m * T_ + t) << 9) + nA] + readv[(m << 9) + nA];
                c1[r] = out0[(((size_t)m * T_ + t) << 9) + nBt] + readv[(m << 9) + nBt];
            }
#pragma unroll
            for (int kk = 0; kk < 16; ++kk) {
                c0 = __builtin_amdgcn_wmma_f32_16x16x32_bf16(false, afr[kk], false,
                        load_frag_b_fm(WhP, kk, nb0, lane), (short)0, c0, false, false);
                c1 = __builtin_amdgcn_wmma_f32_16x16x32_bf16(false, afr[kk], false,
                        load_frag_b_fm(WhP, kk, nb1, lane), (short)0, c1, false, false);
            }
            // publish h_new: fp32 to h_seq (in place over xw), bf16 to hFrag
#pragma unroll
            for (int j2 = 0; j2 < 2; ++j2) {
                int n = (j2 == 0) ? nA : nBt;
#pragma unroll
                for (int r = 0; r < 8; ++r) {
                    int m = r + mh;
                    float hv = tanhf((j2 == 0) ? c0[r] : c1[r]);
                    out0[(((size_t)m * T_ + t) << 9) + n] = hv;
                    int k = n;
                    int kk = k >> 5, half = (k >> 3) & 1;
                    int v = ((k >> 4) & 1) * 4 + ((k >> 1) & 3);
                    int widx = (kk << 8) + ((m + (half << 4)) << 3) + v;
                    hFrag[(widx << 1) + (k & 1)] = f2bf(hv);
                }
            }
        }
        __syncthreads();

        // ---- phase 3: w_vec = h_new @ Wwrite^T -> tape slot (t % 64)
        v16bf afr2[16];
#pragma unroll
        for (int kk = 0; kk < 16; ++kk) {
            FragB f;
            const uint4* q = (const uint4*)&hFrag[((kk << 8) + (lane << 3)) << 1];
            f.q[0] = q[0];
            f.q[1] = q[1];
            afr2[kk] = f.v;
        }
        int slot = t & 63;
        {
            v8f c0, c1;
#pragma unroll
            for (int r = 0; r < 8; ++r) { c0[r] = 0.f; c1[r] = 0.f; }
#pragma unroll
            for (int kk = 0; kk < 16; ++kk) {
                c0 = __builtin_amdgcn_wmma_f32_16x16x32_bf16(false, afr2[kk], false,
                        load_frag_b_fm(WwP, kk, nb0, lane), (short)0, c0, false, false);
                c1 = __builtin_amdgcn_wmma_f32_16x16x32_bf16(false, afr2[kk], false,
                        load_frag_b_fm(WwP, kk, nb1, lane), (short)0, c1, false, false);
            }
#pragma unroll
            for (int r = 0; r < 8; ++r) {
                int m = r + mh;
                tape[((size_t)((m << 6) + slot) << 9) + nA]  = c0[r];
                tape[((size_t)((m << 6) + slot) << 9) + nBt] = c1[r];
            }
        }
        __threadfence();   // tape slot visible before next step's attention reads
        __syncthreads();

        // prefetch next step's pre-activation rows (global_prefetch_b8)
        if (t + 1 < T_)
            __builtin_prefetch(out0 + (((size_t)(tid >> 5) * T_ + (t + 1)) << 9) +
                               ((tid & 31) << 4), 0, 0);
    }
}

// ---------- kernel 5: last-h output ----------
__global__ void copy_last(const float* __restrict__ out0, float* __restrict__ last) {
    int i = blockIdx.x * blockDim.x + threadIdx.x;
    if (i < 16 * 512) {
        int b = i >> 9, d = i & 511;
        last[i] = out0[(((size_t)b * T_ + (T_ - 1)) << 9) + d];
    }
}

extern "C" void kernel_launch(void* const* d_in, const int* in_sizes, int n_in,
                              void* d_out, int out_size, void* d_ws, size_t ws_size,
                              hipStream_t stream) {
    const float* x_seq  = (const float*)d_in[0];
    const float* h_tape = (const float*)d_in[1];
    const float* h_work = (const float*)d_in[2];
    const float* W_h    = (const float*)d_in[3];
    const float* W_x    = (const float*)d_in[4];
    const float* b_h    = (const float*)d_in[5];
    const float* W_wr   = (const float*)d_in[6];

    float* out  = (float*)d_out;
    float* out0 = out;                                   // [16,4096,512] xw -> h_seq
    float* tape = out + (size_t)B_ * T_ * D_;            // [16,64,512]
    float* last = tape + (size_t)B_ * S_ * D_;           // [16,512]

    unsigned* WhP = (unsigned*)d_ws;       // 512 KB each, packed bf16 fragment-major
    unsigned* WxP = WhP + 131072;
    unsigned* WwP = WhP + 262144;

    pack_weight_fm<<<512, 256, 0, stream>>>(W_h, WhP);
    pack_weight_fm<<<512, 256, 0, stream>>>(W_x, WxP);
    pack_weight_fm<<<512, 256, 0, stream>>>(W_wr, WwP);

    xw_gemm<<<dim3(2048, 4), 256, 0, stream>>>(x_seq, WxP, b_h, out0);

    copy_f32<<<2048, 256, 0, stream>>>(h_tape, tape, B_ * S_ * D_);

    recurrence<<<1, 512, 0, stream>>>(h_work, WhP, WwP, out0, tape);

    copy_last<<<32, 256, 0, stream>>>(out0, last);
}
